// SpecificMoE_63702954934785
// MI455X (gfx1250) — matrix operand order repaired and checked by reading, not compiled, not run on hardware
//
#include <hip/hip_runtime.h>
#include <hip/hip_bf16.h>

#define DIM 1024
#define HIDDEN 4096
#define NE 8
#define TOPK 2
#define NTOK 8192            // 4 * 2048 tokens
#define CAP 16384            // worst case: every topk slot goes to one expert
#define MROWS 32             // token rows per block (2 x 16-row WMMA tiles)
#define HHALF 2048           // hidden split: H processed in two halves

// output sections (flat fp32, in reference return order)
#define OUT_LOGITS (NTOK * DIM)                  // 8388608
#define OUT_PROBS  (OUT_LOGITS + NTOK * NE)
#define OUT_IDX    (OUT_PROBS + NTOK * NE)
#define OUT_TOPP   (OUT_IDX + NTOK * TOPK)

// padded LDS row strides (in ushorts) to avoid bank conflicts on b128 A-loads
#define XS_STRIDE 1032       // DIM + 8
#define HS_STRIDE 2056       // HHALF + 8

typedef __attribute__((ext_vector_type(16))) __bf16 v16bf;
typedef __attribute__((ext_vector_type(8)))  float  v8f;

union ABv { v16bf v; uint4 q[2]; };

__device__ __forceinline__ unsigned short f32_to_bf16(float f) {
  union { float f; unsigned int u; } c; c.f = f;
  unsigned int u = c.u;
  u += 0x7FFFu + ((u >> 16) & 1u);   // round to nearest even
  return (unsigned short)(u >> 16);
}

// ---------------------------------------------------------------------------
// zero mixed-output region + expert counters
__global__ void zero_kernel(float* __restrict__ out, int n, int* __restrict__ cnt) {
  int i = blockIdx.x * blockDim.x + threadIdx.x;
  if (i < n) out[i] = 0.0f;
  if (blockIdx.x == 0 && threadIdx.x < NE) cnt[threadIdx.x] = 0;
}

// ---------------------------------------------------------------------------
// fp32 [E][R][C] -> bf16 [E][C][R]  (tiled transpose, coalesced both sides)
__global__ void transpose_convert(const float* __restrict__ in,
                                  unsigned short* __restrict__ out,
                                  int R, int C) {
  __shared__ unsigned short tile[32][33];
  int e = blockIdx.z;
  const float* src = in + (size_t)e * R * C;
  unsigned short* dst = out + (size_t)e * R * C;
  int c0 = blockIdx.x * 32, r0 = blockIdx.y * 32;
#pragma unroll
  for (int i = 0; i < 32; i += 8) {
    int r = r0 + threadIdx.y + i, c = c0 + threadIdx.x;
    tile[threadIdx.y + i][threadIdx.x] = f32_to_bf16(src[(size_t)r * C + c]);
  }
  __syncthreads();
#pragma unroll
  for (int i = 0; i < 32; i += 8) {
    int c = c0 + threadIdx.y + i, r = r0 + threadIdx.x;
    dst[(size_t)c * R + r] = tile[threadIdx.x][threadIdx.y + i];
  }
}

// ---------------------------------------------------------------------------
// router: one wave32 per token. logits -> softmax -> top2; build expert lists.
__global__ __launch_bounds__(256)
void router_kernel(const float* __restrict__ x, const float* __restrict__ Wr,
                   const float* __restrict__ br, float* __restrict__ out,
                   int* __restrict__ cnt, int* __restrict__ etok,
                   float* __restrict__ eprob) {
  int wave = threadIdx.x >> 5, lane = threadIdx.x & 31;
  int t = blockIdx.x * 8 + wave;
  const float* xr = x + (size_t)t * DIM;

  float acc[NE];
#pragma unroll
  for (int e = 0; e < NE; ++e) acc[e] = 0.0f;

  for (int d = lane; d < DIM; d += 32) {
    float xv = xr[d];
    const float4* w = (const float4*)(Wr + d * NE);
    float4 w0 = w[0], w1 = w[1];
    acc[0] += xv * w0.x; acc[1] += xv * w0.y; acc[2] += xv * w0.z; acc[3] += xv * w0.w;
    acc[4] += xv * w1.x; acc[5] += xv * w1.y; acc[6] += xv * w1.z; acc[7] += xv * w1.w;
  }
#pragma unroll
  for (int off = 16; off > 0; off >>= 1)
#pragma unroll
    for (int e = 0; e < NE; ++e) acc[e] += __shfl_xor(acc[e], off, 32);

  if (lane == 0) {
    float logits[NE], probs[NE];
    float m = -1e30f;
#pragma unroll
    for (int e = 0; e < NE; ++e) { logits[e] = acc[e] + br[e]; m = fmaxf(m, logits[e]); }
    float s = 0.0f;
#pragma unroll
    for (int e = 0; e < NE; ++e) { probs[e] = expf(logits[e] - m); s += probs[e]; }
    float inv = 1.0f / s;
#pragma unroll
    for (int e = 0; e < NE; ++e) probs[e] *= inv;

    int i0 = 0;
#pragma unroll
    for (int e = 1; e < NE; ++e) if (probs[e] > probs[i0]) i0 = e;   // lowest idx on tie
    int i1 = (i0 == 0) ? 1 : 0;
#pragma unroll
    for (int e = 0; e < NE; ++e) if (e != i0 && probs[e] > probs[i1]) i1 = e;

    float* lg = out + OUT_LOGITS + t * NE;
    float* pr = out + OUT_PROBS + t * NE;
#pragma unroll
    for (int e = 0; e < NE; ++e) { lg[e] = logits[e]; pr[e] = probs[e]; }
    out[OUT_IDX  + t * 2 + 0] = (float)i0;
    out[OUT_IDX  + t * 2 + 1] = (float)i1;
    out[OUT_TOPP + t * 2 + 0] = probs[i0];
    out[OUT_TOPP + t * 2 + 1] = probs[i1];

    int p0 = atomicAdd(&cnt[i0], 1);
    etok[i0 * CAP + p0] = t; eprob[i0 * CAP + p0] = probs[i0];
    int p1 = atomicAdd(&cnt[i1], 1);
    etok[i1 * CAP + p1] = t; eprob[i1 * CAP + p1] = probs[i1];
  }
}

// ---------------------------------------------------------------------------
// fused expert MLP over a 32-token tile, 2-way M-blocking per wave:
// for each hidden half: H = gelu(X @ W1 + b1) into LDS, then partial
// accumulation of Out = H @ W2 into persistent per-wave register tiles.
__global__ __launch_bounds__(256)
void moe_expert_kernel(const float* __restrict__ x,
                       const unsigned short* __restrict__ W1T,  // [E][H][D] bf16
                       const float* __restrict__ b1,
                       const unsigned short* __restrict__ W2T,  // [E][D][H] bf16
                       const float* __restrict__ b2,
                       const int* __restrict__ cnt,
                       const int* __restrict__ etok,
                       const float* __restrict__ eprob,
                       float* __restrict__ out) {
  extern __shared__ char smem[];
  unsigned short* Xs = (unsigned short*)smem;                // MROWS * XS_STRIDE
  unsigned short* Hs = Xs + MROWS * XS_STRIDE;               // MROWS * HS_STRIDE
  int*   toks = (int*)(Hs + MROWS * HS_STRIDE);
  float* pws  = (float*)(toks + MROWS);

  int e = blockIdx.y;
  int n = cnt[e];
  int base = blockIdx.x * MROWS;
  if (base >= n) return;                            // uniform per block
  int rows = n - base; if (rows > MROWS) rows = MROWS;

  int tid = threadIdx.x;
  if (tid < MROWS) {
    if (tid < rows) {
      toks[tid] = etok[e * CAP + base + tid];
      pws[tid]  = eprob[e * CAP + base + tid];
    } else { toks[tid] = 0; pws[tid] = 0.0f; }
  }
  __syncthreads();

  // gather X rows, convert fp32 -> bf16 into LDS
  for (int idx = tid; idx < MROWS * DIM; idx += 256) {
    int r = idx >> 10, c = idx & (DIM - 1);
    float v = (r < rows) ? x[(size_t)toks[r] * DIM + c] : 0.0f;
    Xs[r * XS_STRIDE + c] = f32_to_bf16(v);
  }
  __syncthreads();

  int lane = tid & 31, wave = tid >> 5;
  int lane16 = lane & 15, laneHalf = lane >> 4;
  // WMMA per-lane layouts (ISA 7.12.2): A row = lane%16, K halves split by lane/16;
  // B col = lane%16, K = 8*(lane/16) + {0..7, 16..23}; C row = r + 8*(lane/16), col = lane%16.

  // persistent output accumulators: 8 N-chunks x 2 M-tiles per wave (128 VGPRs)
  v8f oacc[8][2];
#pragma unroll
  for (int j = 0; j < 8; ++j) {
    oacc[j][0] = (v8f){};
    oacc[j][1] = (v8f){};
  }

  for (int hb = 0; hb < 2; ++hb) {
    // ---- GEMM1 (half): H[:, hb*2048 .. ) = gelu(X @ W1 + b1) ----
    for (int nc = wave; nc < HHALF / 16; nc += 8) {
      int lcol = nc * 16 + lane16;                 // column within this half
      int gcol = hb * HHALF + lcol;                // global hidden column
      const unsigned short* bp = W1T + ((size_t)e * HIDDEN + gcol) * DIM + 8 * laneHalf;
      const unsigned short* a0 = Xs + lane16 * XS_STRIDE + 8 * laneHalf;
      const unsigned short* a1 = a0 + 16 * XS_STRIDE;
      v8f acc0 = {}, acc1 = {};
      for (int k0 = 0; k0 < DIM; k0 += 32) {
        __builtin_prefetch(bp + k0 + 256, 0, 3);
        ABv a, c, b;
        a.q[0] = *(const uint4*)(a0 + k0);
        a.q[1] = *(const uint4*)(a0 + k0 + 16);
        c.q[0] = *(const uint4*)(a1 + k0);
        c.q[1] = *(const uint4*)(a1 + k0 + 16);
        b.q[0] = *(const uint4*)(bp + k0);
        b.q[1] = *(const uint4*)(bp + k0 + 16);
        acc0 = __builtin_amdgcn_wmma_f32_16x16x32_bf16(false, a.v, false, b.v,
                                                       (short)0, acc0, false, false);
        acc1 = __builtin_amdgcn_wmma_f32_16x16x32_bf16(false, c.v, false, b.v,
                                                       (short)0, acc1, false, false);
      }
      float hbias = b1[e * HIDDEN + gcol];
#pragma unroll
      for (int r = 0; r < 8; ++r) {
        int row = r + 8 * laneHalf;
        float v0 = acc0[r] + hbias;
        v0 = 0.5f * v0 * (1.0f + erff(v0 * 0.70710678118654752f));
        Hs[row * HS_STRIDE + lcol] = f32_to_bf16(v0);
        float v1 = acc1[r] + hbias;
        v1 = 0.5f * v1 * (1.0f + erff(v1 * 0.70710678118654752f));
        Hs[(row + 16) * HS_STRIDE + lcol] = f32_to_bf16(v1);
      }
    }
    __syncthreads();

    // ---- GEMM2 (partial K): oacc += H_half @ W2[hb*2048 .. ] ----
#pragma unroll
    for (int j = 0; j < 8; ++j) {
      int col = (wave + j * 8) * 16 + lane16;      // output column
      const unsigned short* bp = W2T + ((size_t)e * DIM + col) * HIDDEN
                                 + hb * HHALF + 8 * laneHalf;
      const unsigned short* a0 = Hs + lane16 * HS_STRIDE + 8 * laneHalf;
      const unsigned short* a1 = a0 + 16 * HS_STRIDE;
      for (int k0 = 0; k0 < HHALF; k0 += 32) {
        __builtin_prefetch(bp + k0 + 256, 0, 3);
        ABv a, c, b;
        a.q[0] = *(const uint4*)(a0 + k0);
        a.q[1] = *(const uint4*)(a0 + k0 + 16);
        c.q[0] = *(const uint4*)(a1 + k0);
        c.q[1] = *(const uint4*)(a1 + k0 + 16);
        b.q[0] = *(const uint4*)(bp + k0);
        b.q[1] = *(const uint4*)(bp + k0 + 16);
        oacc[j][0] = __builtin_amdgcn_wmma_f32_16x16x32_bf16(false, a.v, false, b.v,
                                                             (short)0, oacc[j][0], false, false);
        oacc[j][1] = __builtin_amdgcn_wmma_f32_16x16x32_bf16(false, c.v, false, b.v,
                                                             (short)0, oacc[j][1], false, false);
      }
    }
    __syncthreads();   // Hs reused by next hidden half
  }

  // ---- epilogue: Out[token, col] += p * (acc + b2[col]) ----
#pragma unroll
  for (int j = 0; j < 8; ++j) {
    int col = (wave + j * 8) * 16 + lane16;
    float ob = b2[e * DIM + col];
#pragma unroll
    for (int m = 0; m < 2; ++m) {
#pragma unroll
      for (int r = 0; r < 8; ++r) {
        int row = m * 16 + r + 8 * laneHalf;
        if (row < rows)
          atomicAdd(&out[(size_t)toks[row] * DIM + col],
                    pws[row] * (oacc[j][m][r] + ob));
      }
    }
  }
}

// ---------------------------------------------------------------------------
extern "C" void kernel_launch(void* const* d_in, const int* in_sizes, int n_in,
                              void* d_out, int out_size, void* d_ws, size_t ws_size,
                              hipStream_t stream) {
  const float* x  = (const float*)d_in[0];
  const float* Wr = (const float*)d_in[1];
  const float* br = (const float*)d_in[2];
  const float* W1 = (const float*)d_in[3];
  const float* b1 = (const float*)d_in[4];
  const float* W2 = (const float*)d_in[5];
  const float* b2 = (const float*)d_in[6];
  float* out = (float*)d_out;

  // workspace layout
  char* ws = (char*)d_ws;
  int*            cnt   = (int*)ws;                                   // 8 ints (256B slot)
  int*            etok  = (int*)(ws + 256);                           // NE*CAP ints
  float*          eprob = (float*)(ws + 256 + (size_t)NE * CAP * 4);  // NE*CAP floats
  unsigned short* W1T   = (unsigned short*)(ws + 256 + (size_t)NE * CAP * 8);
  unsigned short* W2T   = W1T + (size_t)NE * DIM * HIDDEN;            // 64MB each

  int mixedN = NTOK * DIM;
  zero_kernel<<<(mixedN + 255) / 256, 256, 0, stream>>>(out, mixedN, cnt);

  dim3 tb(32, 8);
  transpose_convert<<<dim3(HIDDEN / 32, DIM / 32, NE), tb, 0, stream>>>(W1, W1T, DIM, HIDDEN);
  transpose_convert<<<dim3(DIM / 32, HIDDEN / 32, NE), tb, 0, stream>>>(W2, W2T, HIDDEN, DIM);

  router_kernel<<<NTOK / 8, 256, 0, stream>>>(x, Wr, br, out, cnt, etok, eprob);

  size_t shmem = (size_t)(MROWS * XS_STRIDE + MROWS * HS_STRIDE) * sizeof(unsigned short)
               + MROWS * (sizeof(int) + sizeof(float));              // ~194 KB < 320 KB/WGP
  moe_expert_kernel<<<dim3(CAP / MROWS, NE), 256, shmem, stream>>>(
      x, W1T, b1, W2T, b2, cnt, etok, eprob, out);
}
